// AttentionBlock_5746666242430
// MI455X (gfx1250) — matrix operand order repaired and checked by reading, not compiled, not run on hardware
//
#include <hip/hip_runtime.h>

#define B_    4
#define C_    512
#define C8_   64
#define N_    4096     // H*W
#define OUTR_ 640      // 64 q + 64 k + 512 v output rows

typedef __attribute__((ext_vector_type(16))) __bf16 v16bf;
typedef __attribute__((ext_vector_type(8)))  __bf16 v8bf;
typedef __attribute__((ext_vector_type(8)))  float  v8f;
typedef int v4i __attribute__((vector_size(16)));   // matches builtin param type

static __device__ __forceinline__ v8f wmma_bf16(v16bf a, v16bf b, v8f c) {
  // D = A(16x32 bf16) * B(32x16 bf16) + C(16x16 f32)
  return __builtin_amdgcn_wmma_f32_16x16x32_bf16(false, a, false, b, (short)0, c,
                                                 false, false);
}
static __device__ __forceinline__ v16bf cat8(v8bf lo, v8bf hi) {
  return __builtin_shufflevector(lo, hi, 0, 1, 2, 3, 4, 5, 6, 7,
                                 8, 9, 10, 11, 12, 13, 14, 15);
}

// ---- gfx1250 async global->LDS copy (ASYNCcnt-tracked DMA), with fallback ----
#if __has_builtin(__builtin_amdgcn_global_load_async_to_lds_b128)
#define ASYNC_LDS 1
#else
#define ASYNC_LDS 0
#endif

static __device__ __forceinline__ void async_copy16(const void* gsrc, void* ldst) {
#if ASYNC_LDS
  __builtin_amdgcn_global_load_async_to_lds_b128(
      (__attribute__((address_space(1))) v4i*)gsrc,
      (__attribute__((address_space(3))) v4i*)ldst, 0, 0);
#else
  *(uint4*)ldst = *(const uint4*)gsrc;
#endif
}
static __device__ __forceinline__ void wait_async0() {
#if ASYNC_LDS
#if __has_builtin(__builtin_amdgcn_s_wait_asynccnt)
  __builtin_amdgcn_s_wait_asynccnt(0);
#else
  asm volatile("s_wait_asynccnt 0x0" ::: "memory");
#endif
#endif
}

// ---------------------------------------------------------------------------
// Kernel 1: fused QKV projection (bf16 WMMA, f32 accumulate).
//   q[n][o] = Wq[o][:]·x[:][n] + bq   -> stored [N,64] bf16 (K-contig rows)
//   k[n][o] = Wk[o][:]·x[:][n] + bk   -> stored [N,64] bf16
//   v[c][n] = Wv[c][:]·x[:][n] + bv   -> stored [C,N] bf16
// ---------------------------------------------------------------------------
__global__ __launch_bounds__(256) void proj_qkv(
    const float* __restrict__ x,
    const float* __restrict__ Wq, const float* __restrict__ bq,
    const float* __restrict__ Wk, const float* __restrict__ bk,
    const float* __restrict__ Wv, const float* __restrict__ bv,
    __bf16* __restrict__ qo, __bf16* __restrict__ ko, __bf16* __restrict__ vo)
{
  const int lane = threadIdx.x & 31;
  const int wv   = threadIdx.x >> 5;
  const int half = lane >> 4;   // 0: lanes 0-15, 1: lanes 16-31
  const int l16  = lane & 15;

  int bid        = blockIdx.x;
  const int nblk = bid & 31;  bid >>= 5;
  const int mt   = bid % (OUTR_ / 16);
  const int b    = bid / (OUTR_ / 16);

  const int n0 = nblk * 128 + wv * 16;   // 8 waves cover 128 pixels
  const int m0 = mt * 16;                // 16 output rows (never straddles q/k/v)

  const float* W; const float* bias; int mrel;
  if (m0 < 64)       { W = Wq; bias = bq; mrel = m0; }
  else if (m0 < 128) { W = Wk; bias = bk; mrel = m0 - 64; }
  else               { W = Wv; bias = bv; mrel = m0 - 128; }

  const float* xb   = x + (size_t)b * C_ * N_;
  const int    nn   = n0 + l16;                      // B-fragment column
  const float* wrow = W + (size_t)(mrel + l16) * C_; // A-fragment row
  const int kbA = 16 * half;   // A element chunk:  K = kbA + e
  const int kbB = 16 * half;   // B element chunk:  K = kbB + e

  v8f acc = {};
  for (int k0 = 0; k0 < C_; k0 += 32) {
    v16bf a;
#pragma unroll
    for (int j = 0; j < 8; ++j) {
      a[j]     = (__bf16)wrow[k0 + kbA + j];
      a[j + 8] = (__bf16)wrow[k0 + kbA + 8 + j];
    }
    v16bf bm;
#pragma unroll
    for (int e = 0; e < 16; ++e)
      bm[e] = (__bf16)xb[(size_t)(k0 + kbB + e) * N_ + nn];
    acc = wmma_bf16(a, bm, acc);
  }

  // Epilogue: routing hoisted out of the store loop; q/k emit one 16B vector
  // store per lane (rows r+8*half are contiguous in the [N,64] layout).
  const int mloc0 = 8 * half;                  // first D row this lane holds
  if (m0 < 128) {
    __bf16* dst = ((m0 < 64) ? qo : ko) + ((size_t)b * N_ + nn) * C8_ + mrel;
    v8bf ovec;
#pragma unroll
    for (int r = 0; r < 8; ++r)
      ovec[r] = (__bf16)(acc[r] + bias[mrel + mloc0 + r]);
    *(v8bf*)(dst + mloc0) = ovec;
  } else {
    __bf16* dst = vo + (size_t)b * C_ * N_ + nn;
#pragma unroll
    for (int r = 0; r < 8; ++r) {
      const int orel = mrel + mloc0 + r;
      dst[(size_t)orel * N_] = (__bf16)(acc[r] + bias[orel]);
    }
  }
}

// ---------------------------------------------------------------------------
// Kernel 2: streaming attention. Per block: 32 query rows x all 512 channels.
// 8 waves: mg = w&1 (16 rows), cg = w>>1 (128 channels); each wave owns a
// 16x128 f32 accumulator (8 WMMA tiles).
// Pipeline per 32-column tile:
//   issue async DMA of the V tile (32KB) into LDS  [ASYNCcnt]
//   S = Q*K^T (4 WMMAs, K direct from global) + online softmax (overlaps DMA)
//   P -> LDS (D-layout to A-layout conversion)
//   s_wait_asynccnt 0 ; barrier ; 8 PV WMMAs fed from LDS ; barrier
// out = gamma*NS*softmax(QK^T)@V^T + x   (style correction applied later)
// ---------------------------------------------------------------------------
__global__ __launch_bounds__(256) void attn_core(
    const __bf16* __restrict__ qg, const __bf16* __restrict__ kg,
    const __bf16* __restrict__ vg, const float* __restrict__ x,
    const float* __restrict__ gamma_p, const int* __restrict__ ns_p,
    float* __restrict__ out)
{
  __shared__ __align__(16) __bf16 Vt[C_ * 32];          // 32KB V tile
  __shared__ __align__(32) __bf16 Pbuf[2][16 * 32];     // 2KB P tiles

  const int lane = threadIdx.x & 31;
  const int wv   = threadIdx.x >> 5;
  const int mg   = wv & 1;
  const int cg   = wv >> 1;
  const int half = lane >> 4;
  const int l16  = lane & 15;

  const int b     = blockIdx.x >> 7;       // N_/32 = 128 m-tiles per batch
  const int mt    = blockIdx.x & 127;
  const int mbase = mt * 32 + mg * 16;
  const int cbase = cg * 128;

  // Q A-fragments (persistent): rows mbase..mbase+15, K = 0..63 in 2 steps
  v16bf qa[2];
  {
    const __bf16* qrow = qg + ((size_t)b * N_ + (mbase + l16)) * C8_ + 16 * half;
#pragma unroll
    for (int ks = 0; ks < 2; ++ks)
      qa[ks] = cat8(*(const v8bf*)(qrow + ks * 32),
                    *(const v8bf*)(qrow + ks * 32 + 8));
  }

  float vmax[8], vsum[8];
  v8f   oacc[8];
  v8f   zero = {};
#pragma unroll
  for (int r = 0; r < 8; ++r) { vmax[r] = -3.0e38f; vsum[r] = 0.f; oacc[r] = zero; }

  const __bf16* kb_ = kg + (size_t)b * N_ * C8_;
  const __bf16* vb_ = vg + (size_t)b * C_ * N_;

  for (int nt = 0; nt < N_; nt += 32) {
    // ---- issue async DMA: V[0..511][nt..nt+31] -> LDS (8 x 16B per thread) ----
    {
      const char* vbase = (const char*)vb_ + (size_t)nt * 2;
#pragma unroll
      for (int j = 0; j < 8; ++j) {
        const int o   = ((int)threadIdx.x + j * 256) * 16; // flat byte in tile
        const int c   = o >> 6;                            // row (64B rows)
        const int col = o & 63;                            // byte within row
        async_copy16(vbase + (size_t)c * (N_ * 2) + col, (char*)Vt + o);
      }
    }

    // ---- S = Q*K^T tile (16 x 32): B-fragments direct from k[N,64] ----
    v8f s0 = zero, s1 = zero;
    {
      const __bf16* kr0 = kb_ + (size_t)(nt + l16) * C8_ + 16 * half;
      const __bf16* kr1 = kb_ + (size_t)(nt + 16 + l16) * C8_ + 16 * half;
      if (nt + 32 < N_) __builtin_prefetch(kr0 + 32 * C8_, 0, 1);
      s0 = wmma_bf16(qa[0], *(const v16bf*)(kr0), s0);
      s0 = wmma_bf16(qa[1], *(const v16bf*)(kr0 + 32), s0);
      s1 = wmma_bf16(qa[0], *(const v16bf*)(kr1), s1);
      s1 = wmma_bf16(qa[1], *(const v16bf*)(kr1 + 32), s1);
    }

    // ---- online softmax per row (row r lives in one 16-lane half) ----
#pragma unroll
    for (int r = 0; r < 8; ++r) {
      float mx = fmaxf(s0[r], s1[r]);
#pragma unroll
      for (int off = 1; off < 16; off <<= 1)
        mx = fmaxf(mx, __shfl_xor(mx, off, 32));
      const float mnew = fmaxf(vmax[r], mx);
      const float sc   = __expf(vmax[r] - mnew);
      const float p0   = __expf(s0[r] - mnew);
      const float p1   = __expf(s1[r] - mnew);
      s0[r] = p0; s1[r] = p1;
      float ls = p0 + p1;
#pragma unroll
      for (int off = 1; off < 16; off <<= 1)
        ls += __shfl_xor(ls, off, 32);
      vsum[r] = vsum[r] * sc + ls;
      vmax[r] = mnew;
#pragma unroll
      for (int t = 0; t < 8; ++t) oacc[t][r] *= sc;
    }

    // ---- P: D-layout -> LDS row-major (one c-group writes per m-group) ----
    if (cg == 0) {
#pragma unroll
      for (int r = 0; r < 8; ++r) {
        const int ml = r + 8 * half;
        Pbuf[mg][ml * 32 + l16]      = (__bf16)s0[r];
        Pbuf[mg][ml * 32 + 16 + l16] = (__bf16)s1[r];
      }
    }

    wait_async0();        // this wave's V DMA chunks landed in LDS
    __syncthreads();      // all waves' DMA done + P visible

    // ---- O += P*V^T : A from LDS P, B from LDS V tile (K-contig rows) ----
    v16bf pa;
    {
      const __bf16* prow = &Pbuf[mg][l16 * 32 + 16 * half];
      pa = cat8(*(const v8bf*)(prow), *(const v8bf*)(prow + 8));
    }
#pragma unroll
    for (int t = 0; t < 8; ++t) {
      const __bf16* vrow = &Vt[(size_t)(cbase + t * 16 + l16) * 32 + 16 * half];
      oacc[t] = wmma_bf16(pa, *(const v16bf*)(vrow), oacc[t]);
    }
    __syncthreads();      // safe to overwrite Vt / Pbuf next iteration
  }

  // ---- finalize: gamma * NS * O/rowsum + x ----
  const float g = gamma_p[0] * (float)ns_p[0];
#pragma unroll
  for (int r = 0; r < 8; ++r) {
    const int   m    = mbase + r + 8 * half;
    const float coef = g / vsum[r];
#pragma unroll
    for (int t = 0; t < 8; ++t) {
      const int    c   = cbase + t * 16 + l16;
      const size_t idx = ((size_t)b * C_ + c) * N_ + m;
      out[idx] = oacc[t][r] * coef + x[idx];
    }
  }
}

// ---------------------------------------------------------------------------
// Kernel 3: style-mask correction. The mask only zeroes attn[b, 0, sidx], so
// subtract gamma*NS*p[0,sidx]*v[:,sidx] from output column m=0.
// ---------------------------------------------------------------------------
__global__ __launch_bounds__(256) void style_fix(
    const __bf16* __restrict__ qg, const __bf16* __restrict__ kg,
    const __bf16* __restrict__ vg, const float* __restrict__ gamma_p,
    const int* __restrict__ sidx_p, const int* __restrict__ ns_p,
    float* __restrict__ out)
{
  __shared__ float q0[C8_];
  __shared__ float red[256];
  __shared__ float coef_s;

  const int b = blockIdx.x;
  const int t = threadIdx.x;

  const __bf16* qrow = qg + (size_t)b * N_ * C8_;  // row n=0
  if (t < C8_) q0[t] = (float)qrow[t];
  __syncthreads();

  float svals[16];
  float lmax = -3.0e38f;
#pragma unroll 4
  for (int i = 0; i < 16; ++i) {
    const int n = t + i * 256;
    const __bf16* kr = kg + ((size_t)b * N_ + n) * C8_;
    float s = 0.f;
    for (int o = 0; o < C8_; ++o) s += q0[o] * (float)kr[o];
    svals[i] = s;
    lmax = fmaxf(lmax, s);
  }
  red[t] = lmax; __syncthreads();
  for (int st = 128; st > 0; st >>= 1) {
    if (t < st) red[t] = fmaxf(red[t], red[t + st]);
    __syncthreads();
  }
  const float gmax = red[0]; __syncthreads();

  float lsum = 0.f;
#pragma unroll
  for (int i = 0; i < 16; ++i) lsum += __expf(svals[i] - gmax);
  red[t] = lsum; __syncthreads();
  for (int st = 128; st > 0; st >>= 1) {
    if (t < st) red[t] += red[t + st];
    __syncthreads();
  }
  const float gsum = red[0];

  if (t == 0) {
    const int sidx = sidx_p[b];
    const __bf16* kr = kg + ((size_t)b * N_ + sidx) * C8_;
    float s = 0.f;
    for (int o = 0; o < C8_; ++o) s += q0[o] * (float)kr[o];
    coef_s = gamma_p[0] * (float)ns_p[0] * __expf(s - gmax) / gsum;
  }
  __syncthreads();

  const int sidx = sidx_p[b];
  for (int c = t; c < C_; c += 256) {
    const size_t idx = ((size_t)b * C_ + c) * N_;  // m = 0
    out[idx] -= coef_s * (float)vg[((size_t)b * C_ + c) * N_ + sidx];
  }
}

// ---------------------------------------------------------------------------
extern "C" void kernel_launch(void* const* d_in, const int* in_sizes, int n_in,
                              void* d_out, int out_size, void* d_ws, size_t ws_size,
                              hipStream_t stream) {
  (void)in_sizes; (void)n_in; (void)out_size; (void)ws_size;
  const float* x  = (const float*)d_in[0];
  const float* Wq = (const float*)d_in[1];
  const float* bq = (const float*)d_in[2];
  const float* Wk = (const float*)d_in[3];
  const float* bk = (const float*)d_in[4];
  const float* Wv = (const float*)d_in[5];
  const float* bv = (const float*)d_in[6];
  const float* gm = (const float*)d_in[7];
  const int*   si = (const int*)d_in[8];
  const int*   ns = (const int*)d_in[9];
  float* out = (float*)d_out;

  // workspace: q [B,N,64] bf16 (2MB) | k [B,N,64] bf16 (2MB) | v [B,C,N] bf16 (16MB)
  __bf16* wq  = reinterpret_cast<__bf16*>(d_ws);
  __bf16* wk  = wq + (size_t)B_ * N_ * C8_;
  __bf16* wvv = wk + (size_t)B_ * N_ * C8_;

  proj_qkv<<<dim3(B_ * (OUTR_ / 16) * (N_ / 128)), dim3(256), 0, stream>>>(
      x, Wq, bq, Wk, bk, Wv, bv, wq, wk, wvv);
  attn_core<<<dim3(B_ * (N_ / 32)), dim3(256), 0, stream>>>(
      wq, wk, wvv, x, gm, ns, out);
  style_fix<<<dim3(B_), dim3(256), 0, stream>>>(wq, wk, wvv, gm, si, ns, out);
}
